// Glm4MoE_85255100825929
// MI455X (gfx1250) — matrix operand order repaired and checked by reading, not compiled, not run on hardware
//
#include <hip/hip_runtime.h>

// ---------------- problem constants ----------------
constexpr int Bc = 1;
constexpr int S  = 2048;
constexpr int D  = 1024;
constexpr int E  = 64;
constexpr int K  = 8;
constexpr int F  = 768;
constexpr int SF = 768;
constexpr int C  = 512;
constexpr int T  = Bc * S;          // 2048 tokens
constexpr int N  = T * K;           // 16384 assignments
constexpr float ROUTED_SCALE = 1.0f;

// ---------------- types ----------------
using bf16_t = unsigned short;
typedef __bf16        v16bf __attribute__((ext_vector_type(16)));
typedef float         v8f   __attribute__((ext_vector_type(8)));
typedef unsigned int  u32x4 __attribute__((ext_vector_type(4)));
typedef unsigned int  u32x2 __attribute__((ext_vector_type(2)));
typedef float         f32x4 __attribute__((ext_vector_type(4)));

union BF16x16 { v16bf v; u32x4 q[2]; };

__device__ __forceinline__ bf16_t f2bf(float f) {
  unsigned int u = __float_as_uint(f);
  unsigned int r = u + 0x7FFFu + ((u >> 16) & 1u);   // round-to-nearest-even
  return (bf16_t)(r >> 16);
}

// ---------------- small kernels ----------------
__global__ void xbf_kernel(const float* __restrict__ x, bf16_t* __restrict__ xbf, int n) {
  int i = blockIdx.x * 256 + threadIdx.x;
  if (i < n) xbf[i] = f2bf(x[i]);
}

__global__ void init_slots_kernel(int* __restrict__ slot_to, int n) {
  int i = blockIdx.x * 256 + threadIdx.x;
  if (i < n) slot_to[i] = -1;
}

// one wave (32 lanes) per token; each lane owns experts {lane, lane+32}
__global__ void __launch_bounds__(256) router_kernel(const float* __restrict__ x,
                                                     const float* __restrict__ rw,
                                                     int* __restrict__ flat_e,
                                                     float* __restrict__ flat_w) {
  const int lane = threadIdx.x & 31;
  const int wv   = threadIdx.x >> 5;
  const int t    = blockIdx.x * 8 + wv;
  if (t >= T) return;

  const f32x4* xr = (const f32x4*)(x + (size_t)t * D);
  const f32x4* r0 = (const f32x4*)(rw + (size_t)lane * D);
  const f32x4* r1 = (const f32x4*)(rw + (size_t)(lane + 32) * D);
  float a0 = 0.f, a1 = 0.f;
  for (int i = 0; i < D / 4; ++i) {
    f32x4 xv = xr[i], v0 = r0[i], v1 = r1[i];
    a0 += xv.x * v0.x + xv.y * v0.y + xv.z * v0.z + xv.w * v0.w;
    a1 += xv.x * v1.x + xv.y * v1.y + xv.z * v1.z + xv.w * v1.w;
  }
  // softmax over 64 logits
  float m = fmaxf(a0, a1);
  for (int off = 16; off; off >>= 1) m = fmaxf(m, __shfl_xor(m, off, 32));
  float p0 = __expf(a0 - m), p1 = __expf(a1 - m);
  float s = p0 + p1;
  for (int off = 16; off; off >>= 1) s += __shfl_xor(s, off, 32);
  p0 /= s; p1 /= s;

  // iterative top-8 (argmax with first-index tie-break)
  float selw[K]; int seli[K]; float wsum = 0.f;
  for (int k = 0; k < K; ++k) {
    float v; int id;
    if (p0 >= p1) { v = p0; id = lane; } else { v = p1; id = lane + 32; }
    for (int off = 16; off; off >>= 1) {
      float ov = __shfl_xor(v, off, 32);
      int   oi = __shfl_xor(id, off, 32);
      if (ov > v || (ov == v && oi < id)) { v = ov; id = oi; }
    }
    selw[k] = v; seli[k] = id; wsum += v;
    if (id == lane)           p0 = -1.f;
    else if (id == lane + 32) p1 = -1.f;
  }
  if (lane == 0) {
    for (int k = 0; k < K; ++k) {
      flat_e[t * K + k] = seli[k];
      flat_w[t * K + k] = selw[k] / wsum;
    }
  }
}

// deterministic, order-preserving capacity assignment (matches reference cumsum)
__global__ void assign_kernel(const int* __restrict__ flat_e,
                              int* __restrict__ slot_of,
                              int* __restrict__ slot_to) {
  const int e = threadIdx.x;      // 64 threads, one per expert
  int cnt = 0;
  for (int i = 0; i < N; ++i) {
    int fe = flat_e[i];
    if (fe == e) {
      if (cnt < C) {
        int sidx = e * C + cnt;
        slot_of[i] = sidx;
        slot_to[sidx] = i;
      } else {
        slot_of[i] = -1;          // dropped (over capacity)
      }
      cnt++;
    }
  }
}

// fill expert capacity buffers (bf16), zero for empty slots
__global__ void __launch_bounds__(256) dispatch_kernel(const float* __restrict__ x,
                                                       const int* __restrict__ slot_to,
                                                       bf16_t* __restrict__ xb) {
  const int slot = blockIdx.x;
  const int a = slot_to[slot];
  bf16_t* dst = xb + (size_t)slot * D;
  if (a < 0) {
    for (int d = threadIdx.x; d < D; d += 256) dst[d] = 0;
  } else {
    const float* src = x + (size_t)(a / K) * D;
    for (int d = threadIdx.x; d < D; d += 256) dst[d] = f2bf(src[d]);
  }
}

// ---------------- WMMA GEMM ----------------
// C_out(M x Nn) = A(M x K) * B(K x Nn), A in bf16, B in fp32 (converted inline).
// Block tile 128x128, 8 waves in 2(M) x 4(N): wave tile 64x32 = 4x2 wmma frags.
// A-tile staged with async global->LDS copies (ASYNCcnt); B panels converted
// fp32->bf16 by the VALU while the async engine moves A.
// GATEUP=1: two B matrices, epilogue h = silu(g)*u written as bf16.
// GATEUP=0: single B, epilogue writes f32.
template <int GATEUP>
__global__ void __launch_bounds__(256) gemm_bf16(
    const bf16_t* __restrict__ A, long long aStride, int lda,
    const float* __restrict__ B0, const float* __restrict__ B1,
    long long bStride, int ldb,
    void* __restrict__ OutP, long long oStride, int ldo, int Kdim) {
  __shared__ bf16_t As[128][72];
  __shared__ bf16_t B0s[128][72];                 // stored transposed: [n][k]
  __shared__ bf16_t B1s[GATEUP ? 128 : 1][72];

  const int tid  = threadIdx.x;
  const int lane = tid & 31;
  const int wv   = tid >> 5;
  const int wm   = wv & 1;        // 0..1 -> M half of block tile
  const int wn   = wv >> 1;       // 0..3 -> N quarter
  const int mBase = blockIdx.y * 128;
  const int nBase = blockIdx.x * 128;
  const int z = blockIdx.z;

  A  += (size_t)z * aStride;
  B0 += (size_t)z * bStride;
  if (GATEUP) B1 += (size_t)z * bStride;

  v8f acc0[4][2], acc1[4][2];
  for (int mi = 0; mi < 4; ++mi)
    for (int ni = 0; ni < 2; ++ni) { acc0[mi][ni] = (v8f)0.f; acc1[mi][ni] = (v8f)0.f; }

  const int half = lane >> 4;     // 0 | 1
  const int ml   = lane & 15;

  for (int kt = 0; kt < Kdim; kt += 64) {
    __syncthreads();
    // --- stage A tile asynchronously: 128 x 64 bf16 = 16KB, 4 x b128/thread ---
    {
      const int c8   = (tid & 7) * 8;   // ushort column chunk (16B)
      const int row0 = tid >> 3;        // 0..31
      for (int i = 0; i < 4; ++i) {
        int row = row0 + i * 32;
        unsigned lds = (unsigned)(size_t)&As[row][c8];
        const bf16_t* gp = A + (size_t)(mBase + row) * lda + kt + c8;
        asm volatile("global_load_async_to_lds_b128 %0, %1, off"
                     :: "v"(lds), "v"(gp) : "memory");
      }
    }
    // --- stage B tile(s): 64 x 128 fp32 -> bf16, transposed into LDS ---
    {
      const int ng = (tid & 31) * 4;
      const int kr = tid >> 5;
      for (int i = 0; i < 8; ++i) {
        int k = kr + i * 8;
        f32x4 v = *(const f32x4*)(B0 + (size_t)(kt + k) * ldb + nBase + ng);
        B0s[ng + 0][k] = f2bf(v.x);
        B0s[ng + 1][k] = f2bf(v.y);
        B0s[ng + 2][k] = f2bf(v.z);
        B0s[ng + 3][k] = f2bf(v.w);
        if constexpr (GATEUP) {
          f32x4 v1 = *(const f32x4*)(B1 + (size_t)(kt + k) * ldb + nBase + ng);
          B1s[ng + 0][k] = f2bf(v1.x);
          B1s[ng + 1][k] = f2bf(v1.y);
          B1s[ng + 2][k] = f2bf(v1.z);
          B1s[ng + 3][k] = f2bf(v1.w);
        }
      }
      // prefetch next K-tile's B panel(s): one 128B line per thread covers 32KB
      if (kt + 64 < Kdim) {
        const int pk = tid >> 2;            // 0..63 rows
        const int pc = (tid & 3) * 32;      // 4 x 32-float (128B) chunks
        __builtin_prefetch(B0 + (size_t)(kt + 64 + pk) * ldb + nBase + pc, 0, 1);
        if constexpr (GATEUP)
          __builtin_prefetch(B1 + (size_t)(kt + 64 + pk) * ldb + nBase + pc, 0, 1);
      }
    }
    asm volatile("s_wait_asynccnt 0" ::: "memory");
    __syncthreads();

    for (int ks = 0; ks < 64; ks += 32) {
      BF16x16 af[4], bf0[2], bf1[2];
      // A fragments (16x32 bf16: lo-lanes K 0..15, hi-lanes K 16..31 of window)
      for (int mi = 0; mi < 4; ++mi) {
        const bf16_t* ap = &As[wm * 64 + mi * 16 + ml][ks + half * 16];
        af[mi].q[0] = *(const u32x4*)ap;
        af[mi].q[1] = *(const u32x4*)(ap + 8);
      }
      // B fragments (32x16 bf16: lo-lanes K 0..15, hi-lanes K 16..31 of window)
      for (int ni = 0; ni < 2; ++ni) {
        const bf16_t* bp = &B0s[wn * 32 + ni * 16 + ml][ks + half * 16];
        bf0[ni].q[0] = *(const u32x4*)bp;
        bf0[ni].q[1] = *(const u32x4*)(bp + 8);
        if constexpr (GATEUP) {
          const bf16_t* bp1 = &B1s[wn * 32 + ni * 16 + ml][ks + half * 16];
          bf1[ni].q[0] = *(const u32x4*)bp1;
          bf1[ni].q[1] = *(const u32x4*)(bp1 + 8);
        }
      }
      for (int mi = 0; mi < 4; ++mi)
        for (int ni = 0; ni < 2; ++ni) {
          acc0[mi][ni] = __builtin_amdgcn_wmma_f32_16x16x32_bf16(
              false, af[mi].v, false, bf0[ni].v, (short)0, acc0[mi][ni], false, false);
          if constexpr (GATEUP)
            acc1[mi][ni] = __builtin_amdgcn_wmma_f32_16x16x32_bf16(
                false, af[mi].v, false, bf1[ni].v, (short)0, acc1[mi][ni], false, false);
        }
    }
  }

  // epilogue (C/D layout: vgpr r -> M = r + 8*half, N = lane&15)
  bf16_t* Oh = (bf16_t*)OutP + (size_t)z * oStride;
  float*  Of = (float*)OutP + (size_t)z * oStride;
  for (int mi = 0; mi < 4; ++mi)
    for (int ni = 0; ni < 2; ++ni) {
      const int rbase = mBase + wm * 64 + mi * 16 + half * 8;
      const int col   = nBase + wn * 32 + ni * 16 + ml;
      for (int r = 0; r < 8; ++r) {
        const int row = rbase + r;
        if constexpr (GATEUP) {
          float g = acc0[mi][ni][r];
          float u = acc1[mi][ni][r];
          float hv = (g / (1.f + __expf(-g))) * u;   // silu(g) * u
          Oh[(size_t)row * ldo + col] = f2bf(hv);
        } else {
          Of[(size_t)row * ldo + col] = acc0[mi][ni][r];
        }
      }
    }
}

// out[t,d] (already holds shared-expert result) += scale * sum_k w * y[slot]
__global__ void __launch_bounds__(256) combine_kernel(const float* __restrict__ y,
                                                      const int* __restrict__ slot_of,
                                                      const float* __restrict__ flat_w,
                                                      float* __restrict__ out) {
  const int idx = blockIdx.x * 256 + threadIdx.x;     // t*D + d
  const int t = idx >> 10;                            // D = 1024
  const int d = idx & 1023;
  float acc = out[idx];
  for (int k = 0; k < K; ++k) {
    int sidx = slot_of[t * K + k];
    if (sidx >= 0)
      acc += ROUTED_SCALE * flat_w[t * K + k] * y[(size_t)sidx * D + d];
  }
  out[idx] = acc;
}

// ---------------- launcher ----------------
static inline size_t align256(size_t v) { return (v + 255) & ~(size_t)255; }

extern "C" void kernel_launch(void* const* d_in, const int* in_sizes, int n_in,
                              void* d_out, int out_size, void* d_ws, size_t ws_size,
                              hipStream_t stream) {
  (void)in_sizes; (void)n_in; (void)out_size; (void)ws_size;
  const float* x       = (const float*)d_in[0];   // (T, D)
  const float* rw      = (const float*)d_in[1];   // (E, D)
  const float* w_gate  = (const float*)d_in[2];   // (E, D, F)
  const float* w_up    = (const float*)d_in[3];   // (E, D, F)
  const float* w_down  = (const float*)d_in[4];   // (E, F, D)
  const float* ws_gu   = (const float*)d_in[5];   // (D, 2*SF)
  const float* ws_dn   = (const float*)d_in[6];   // (SF, D)
  float* out = (float*)d_out;                     // (T, D)

  // workspace carving (~260 MB total)
  char* w = (char*)d_ws;
  size_t off = 0;
  bf16_t* xbf    = (bf16_t*)(w + off); off = align256(off + (size_t)T * D * sizeof(bf16_t));
  int*    flat_e = (int*)   (w + off); off = align256(off + (size_t)N * sizeof(int));
  float*  flat_w = (float*) (w + off); off = align256(off + (size_t)N * sizeof(float));
  int*    slot_of= (int*)   (w + off); off = align256(off + (size_t)N * sizeof(int));
  int*    slot_to= (int*)   (w + off); off = align256(off + (size_t)E * C * sizeof(int));
  bf16_t* xb     = (bf16_t*)(w + off); off = align256(off + (size_t)E * C * D * sizeof(bf16_t));
  bf16_t* hbuf   = (bf16_t*)(w + off); off = align256(off + (size_t)E * C * F * sizeof(bf16_t));
  bf16_t* hs     = (bf16_t*)(w + off); off = align256(off + (size_t)T * SF * sizeof(bf16_t));
  float*  ybuf   = (float*) (w + off); off = align256(off + (size_t)E * C * D * sizeof(float));

  // 1) x -> bf16 copy (for shared-expert GEMM A)
  xbf_kernel<<<(T * D) / 256, 256, 0, stream>>>(x, xbf, T * D);
  // 2) router: logits/softmax/top-8
  router_kernel<<<T / 8, 256, 0, stream>>>(x, rw, flat_e, flat_w);
  // 3) slot map init + deterministic capacity assignment
  init_slots_kernel<<<(E * C) / 256, 256, 0, stream>>>(slot_to, E * C);
  assign_kernel<<<1, 64, 0, stream>>>(flat_e, slot_of, slot_to);
  // 4) dispatch tokens into (E, C, D) bf16 capacity buffers
  dispatch_kernel<<<E * C, 256, 0, stream>>>(x, slot_to, xb);
  // 5) expert fused gate/up + silu  -> h (E, C, F) bf16
  gemm_bf16<1><<<dim3(F / 128, C / 128, E), 256, 0, stream>>>(
      xb, (long long)C * D, D, w_gate, w_up, (long long)D * F, F,
      hbuf, (long long)C * F, F, D);
  // 6) expert down  -> y (E, C, D) fp32
  gemm_bf16<0><<<dim3(D / 128, C / 128, E), 256, 0, stream>>>(
      hbuf, (long long)C * F, F, w_down, nullptr, (long long)F * D, D,
      ybuf, (long long)C * D, D, F);
  // 7) shared expert fused gate/up + silu -> hs (T, SF) bf16
  gemm_bf16<1><<<dim3(SF / 128, T / 128, 1), 256, 0, stream>>>(
      xbf, 0, D, ws_gu, ws_gu + SF, 0, 2 * SF,
      hs, 0, SF, D);
  // 8) shared expert down -> out (T, D) fp32 (full overwrite)
  gemm_bf16<0><<<dim3(D / 128, T / 128, 1), 256, 0, stream>>>(
      hs, 0, SF, ws_dn, nullptr, 0, D,
      out, 0, D, SF);
  // 9) out += routed contributions (deterministic gather, no atomics)
  combine_kernel<<<(T * D) / 256, 256, 0, stream>>>(ybuf, slot_of, flat_w, out);
}
